// Attention_721554506338
// MI455X (gfx1250) — matrix operand order repaired and checked by reading, not compiled, not run on hardware
//
#include <hip/hip_runtime.h>

// Problem constants (from reference)
#define B_  16
#define C_  512
#define T_  1024
#define NH  8
#define HD  64

typedef __attribute__((ext_vector_type(16))) __bf16 v16bf;
typedef __attribute__((ext_vector_type(8)))  float  v8f;

__device__ __forceinline__ unsigned short f2bf(float f) {
  unsigned int u = __float_as_uint(f);
  u += 0x7FFFu + ((u >> 16) & 1u);            // round-to-nearest-even
  return (unsigned short)(u >> 16);
}

union FragU { uint4 q[2]; v16bf v; unsigned short s[16]; };

// A-fragment: 16x32 bf16, row-major source, ld in elements.
// lane<16 holds row m=lane, K=0..7 & 16..23 ; lane>=16 holds K=8..15 & 24..31
__device__ __forceinline__ v16bf load_a(const unsigned short* src, int ld) {
  const int lane = threadIdx.x & 31;
  const int m  = lane & 15;
  const int ko = (lane >> 4) << 3;
  const uint4* p = reinterpret_cast<const uint4*>(src + (size_t)m * ld + ko);
  FragU u; u.q[0] = p[0]; u.q[1] = p[2];
  return u.v;
}

// B-fragment: 32x16 bf16 from "N-major" storage (row n holds its 32 K values
// contiguously), ld = elements between consecutive n.
// lane<16: col n=lane, K=0..7 & 16..23 ; lane>=16: K=8..15 & 24..31
__device__ __forceinline__ v16bf load_b(const unsigned short* src, int ld) {
  const int lane = threadIdx.x & 31;
  const int n  = lane & 15;
  const int ko = (lane >> 4) << 3;
  const uint4* p = reinterpret_cast<const uint4*>(src + (size_t)n * ld + ko);
  FragU u; u.q[0] = p[0]; u.q[1] = p[2];
  return u.v;
}

__device__ __forceinline__ v8f wmma_bf16(v16bf a, v16bf b, v8f c) {
  return __builtin_amdgcn_wmma_f32_16x16x32_bf16(false, a, false, b, (short)0, c,
                                                 false, false);
}

// gfx1250 async copy: 16B global -> LDS per lane, tracked by ASYNCcnt.
// VDST (first operand) = VGPR holding LDS byte address, per VGLOBAL encoding.
__device__ __forceinline__ void async_ld_b128(unsigned lds_off, const void* gptr) {
  asm volatile("global_load_async_to_lds_b128 %0, %1, off"
               :: "v"(lds_off), "v"(gptr) : "memory");
}

// Software-pipelined 16x64 GEMM tile over K=512 (bf16 WMMA, fp32 acc).
struct Acc4 { v8f a0, a1, a2, a3; };
__device__ __forceinline__ Acc4 gemm16x64_k512(const unsigned short* Arow,
                                               const unsigned short* Bb) {
  Acc4 acc; acc.a0 = {}; acc.a1 = {}; acc.a2 = {}; acc.a3 = {};
  v16bf ac  = load_a(Arow, C_);
  v16bf bc0 = load_b(Bb + (size_t)0  * C_, C_);
  v16bf bc1 = load_b(Bb + (size_t)16 * C_, C_);
  v16bf bc2 = load_b(Bb + (size_t)32 * C_, C_);
  v16bf bc3 = load_b(Bb + (size_t)48 * C_, C_);
  for (int kk = 32; kk < C_; kk += 32) {
    // prefetch next k-step while current WMMAs issue
    v16bf an  = load_a(Arow + kk, C_);
    v16bf bn0 = load_b(Bb + (size_t)0  * C_ + kk, C_);
    v16bf bn1 = load_b(Bb + (size_t)16 * C_ + kk, C_);
    v16bf bn2 = load_b(Bb + (size_t)32 * C_ + kk, C_);
    v16bf bn3 = load_b(Bb + (size_t)48 * C_ + kk, C_);
    acc.a0 = wmma_bf16(ac, bc0, acc.a0);
    acc.a1 = wmma_bf16(ac, bc1, acc.a1);
    acc.a2 = wmma_bf16(ac, bc2, acc.a2);
    acc.a3 = wmma_bf16(ac, bc3, acc.a3);
    ac = an; bc0 = bn0; bc1 = bn1; bc2 = bn2; bc3 = bn3;
  }
  acc.a0 = wmma_bf16(ac, bc0, acc.a0);
  acc.a1 = wmma_bf16(ac, bc1, acc.a1);
  acc.a2 = wmma_bf16(ac, bc2, acc.a2);
  acc.a3 = wmma_bf16(ac, bc3, acc.a3);
  return acc;
}

// ---------------------------------------------------------------------------
// Kernel 0: fp32 -> bf16 conversions. x is transposed to token-major.
// ---------------------------------------------------------------------------
__global__ void __launch_bounds__(256) prep_kernel(
    const float* __restrict__ x, const float* __restrict__ wqkv,
    const float* __restrict__ wo,
    unsigned short* __restrict__ Xt, unsigned short* __restrict__ Wq,
    unsigned short* __restrict__ Wob)
{
  const long long NX = (long long)B_ * C_ * T_;        // 8388608
  const long long NW = 3LL * C_ * C_;                  // 786432
  long long i = (long long)blockIdx.x * 256 + threadIdx.x;
  if (i < NX) {
    // i indexes Xt[b][t][c]
    long long c = i & (C_ - 1), t = (i >> 9) & (T_ - 1), b = i >> 19;
    Xt[i] = f2bf(x[(b * C_ + c) * T_ + t]);
  } else if (i < NX + NW) {
    long long j = i - NX; Wq[j] = f2bf(wqkv[j]);
  } else if (i < NX + NW + (long long)C_ * C_) {
    long long j = i - NX - NW; Wob[j] = f2bf(wo[j]);
  }
}

// ---------------------------------------------------------------------------
// Kernel 1: qkv = w_qkv @ X + b, split heads, SO(2) rotate, store bf16:
//   Q (pre-scaled 1/8) and K as [b][nh][t][hd] (token-major)
//   V as [b][nh][hd][t] (hd-major: B-fragment layout for P@V)
// ---------------------------------------------------------------------------
__global__ void __launch_bounds__(128) qkv_kernel(
    const unsigned short* __restrict__ Wq, const unsigned short* __restrict__ Xt,
    const float* __restrict__ bqkv,
    const float* __restrict__ mat_q, const float* __restrict__ mat_k,
    const float* __restrict__ mat_v,
    unsigned short* __restrict__ Qb, unsigned short* __restrict__ Kb,
    unsigned short* __restrict__ Vt)
{
  const int warp = threadIdx.x >> 5;
  const int lane = threadIdx.x & 31;
  const int wid  = blockIdx.x * 4 + warp;        // 24576 waves = 16 * 96 * 16
  const int ntile = wid & 15;
  const int mtile = (wid >> 4) % 96;
  const int b     = wid / (16 * 96);
  const int o0   = mtile * 16;
  const int tok0 = ntile * 64;

  Acc4 acc = gemm16x64_k512(Wq + (size_t)o0 * C_,
                            Xt + ((size_t)b * T_ + tok0) * C_);
  v8f av[4] = {acc.a0, acc.a1, acc.a2, acc.a3};

  const int n   = lane & 15;
  const int mb  = (lane >> 4) << 3;
  const int sec = o0 >> 9;            // 0=q 1=k 2=v
  const int ch0 = o0 & (C_ - 1);
  const int f   = ch0 >> 4;           // rotation-pair index, uniform per tile
  const float* mat = (sec == 0) ? mat_q : (sec == 1) ? mat_k : mat_v;

  float bv[8];
#pragma unroll
  for (int r = 0; r < 8; ++r) bv[r] = bqkv[o0 + mb + r];

#pragma unroll
  for (int j = 0; j < 4; ++j) {
    const int t = tok0 + j * 16 + n;
    const float4 mm = *reinterpret_cast<const float4*>(mat + ((size_t)t * 32 + f) * 4);
#pragma unroll
    for (int r = 0; r < 8; ++r) {
      float own  = av[j][r] + bv[r];
      float part = __shfl_xor(own, 16);        // pair channel sits 8 rows away
      float rot  = (lane < 16) ? (mm.x * own + mm.y * part)    // even comp
                               : (mm.z * part + mm.w * own);   // odd comp
      const int ch = ch0 + mb + r;
      const int c = ch >> 3, h = ch & 7;       // channel = c*8 + head
      if (sec == 0)
        Qb[(((size_t)(b * NH + h)) * T_ + t) * HD + c] = f2bf(rot * 0.125f);
      else if (sec == 1)
        Kb[(((size_t)(b * NH + h)) * T_ + t) * HD + c] = f2bf(rot);
      else
        Vt[(((size_t)(b * NH + h)) * HD + c) * T_ + t] = f2bf(rot);
    }
  }
}

// ---------------------------------------------------------------------------
// Kernel 2: flash attention. One block = one (b, head), 4 waves = 4 query
// tiles of 16. 32-key K/V chunks are cooperatively staged into LDS with
// gfx1250 async-to-LDS copies (double-buffered, ASYNCcnt), shared by all
// 4 waves. Online softmax; P relayout via per-wave LDS slice; output rotated
// and stored token-major bf16 [b][t][c*8+nh] (B-frag layout for kernel 3).
// ---------------------------------------------------------------------------
__global__ void __launch_bounds__(128) attn_kernel(
    const unsigned short* __restrict__ Qb, const unsigned short* __restrict__ Kb,
    const unsigned short* __restrict__ Vt, const float* __restrict__ mat_o,
    unsigned short* __restrict__ Ot)
{
  __shared__ __align__(16) unsigned short Kch[2][32 * 64];  // [key][hd], 4 KB ea
  __shared__ __align__(16) unsigned short Vch[2][64 * 32];  // [hd][key], 4 KB ea
  __shared__ __align__(16) unsigned short Pl[4][16 * 32];   // per-wave P slice

  const int warp = threadIdx.x >> 5;
  const int lane = threadIdx.x & 31;
  const int tid  = threadIdx.x;
  const int grp = blockIdx.x & 15;          // 2048 blocks = 16(b)*8(nh)*16(grp)
  const int nh  = (blockIdx.x >> 4) & 7;
  const int b   = blockIdx.x >> 7;
  const int t0  = (grp * 4 + warp) * 16;
  const int bh  = b * NH + nh;

  const unsigned short* Qbase = Qb + ((size_t)bh * T_ + t0) * HD;
  const unsigned short* Kbase = Kb + (size_t)bh * T_ * HD;
  const unsigned short* Vbase = Vt + (size_t)bh * HD * T_;
  unsigned short* P = Pl[warp];

  // Cooperative async stage of one 32-key chunk into LDS buffer `buf`.
  auto stage = [&](int buf, int key0) {
    // K chunk: 32 rows x 64 hd = 4 KB, contiguous in global. 256 x 16B.
#pragma unroll
    for (int c = 0; c < 2; ++c) {
      const int idx = tid + c * 128;
      async_ld_b128((unsigned)(size_t)&Kch[buf][idx * 8],
                    Kbase + (size_t)key0 * HD + idx * 8);
    }
    // V chunk: 64 hd-rows x 32 keys; each row 64B contiguous (stride T_).
#pragma unroll
    for (int c = 0; c < 2; ++c) {
      const int idx = tid + c * 128;
      const int hd = idx >> 2, q4 = idx & 3;
      async_ld_b128((unsigned)(size_t)&Vch[buf][hd * 32 + q4 * 8],
                    Vbase + (size_t)hd * T_ + key0 + q4 * 8);
    }
  };

  const v16bf qa0 = load_a(Qbase, HD);
  const v16bf qa1 = load_a(Qbase + 32, HD);

  const int n  = lane & 15;
  const int mb = (lane >> 4) << 3;

  float mrow[8], lrow[8];
  v8f oacc[4] = {};
#pragma unroll
  for (int r = 0; r < 8; ++r) { mrow[r] = -3.0e38f; lrow[r] = 0.f; }

  stage(0, 0);                                   // prologue prefetch
  for (int it = 0; it < 32; ++it) {
    const int buf = it & 1;
    asm volatile("s_wait_asynccnt 0" ::: "memory");  // this buf's DMA done
    __syncthreads();                                 // all waves' parts done
    if (it + 1 < 32) stage(buf ^ 1, (it + 1) * 32);  // overlap next DMA

    const unsigned short* Kc = Kch[buf];
    const unsigned short* Vc = Vch[buf];

    v8f s0 = {}, s1 = {};
    s0 = wmma_bf16(qa0, load_b(Kc, 64), s0);               // keys 0..15
    s0 = wmma_bf16(qa1, load_b(Kc + 32, 64), s0);
    s1 = wmma_bf16(qa0, load_b(Kc + 16 * 64, 64), s1);     // keys 16..31
    s1 = wmma_bf16(qa1, load_b(Kc + 16 * 64 + 32, 64), s1);

#pragma unroll
    for (int r = 0; r < 8; ++r) {
      float mx = fmaxf(s0[r], s1[r]);
      mx = fmaxf(mx, __shfl_xor(mx, 1));
      mx = fmaxf(mx, __shfl_xor(mx, 2));
      mx = fmaxf(mx, __shfl_xor(mx, 4));
      mx = fmaxf(mx, __shfl_xor(mx, 8));           // row max across 16 lanes
      const float mnew = fmaxf(mrow[r], mx);
      const float sc = __expf(mrow[r] - mnew);
      const float p0 = __expf(s0[r] - mnew);
      const float p1 = __expf(s1[r] - mnew);
      float rs = p0 + p1;
      rs += __shfl_xor(rs, 1);
      rs += __shfl_xor(rs, 2);
      rs += __shfl_xor(rs, 4);
      rs += __shfl_xor(rs, 8);                     // row sum
      lrow[r] = lrow[r] * sc + rs;
      mrow[r] = mnew;
      oacc[0][r] *= sc; oacc[1][r] *= sc; oacc[2][r] *= sc; oacc[3][r] *= sc;
      P[(mb + r) * 32 + n]      = f2bf(p0);        // relayout C-frag -> A-frag
      P[(mb + r) * 32 + 16 + n] = f2bf(p1);
    }
    asm volatile("s_wait_dscnt 0" ::: "memory");   // wave-local LDS RAW fence
    FragU up;
    const int mm2 = (lane & 15) * 32 + ((lane >> 4) << 3);
#pragma unroll
    for (int e = 0; e < 8; ++e) {
      up.s[e]     = P[mm2 + e];
      up.s[8 + e] = P[mm2 + 16 + e];
    }
    oacc[0] = wmma_bf16(up.v, load_b(Vc + 0 * 16 * 32, 32), oacc[0]);
    oacc[1] = wmma_bf16(up.v, load_b(Vc + 1 * 16 * 32, 32), oacc[1]);
    oacc[2] = wmma_bf16(up.v, load_b(Vc + 2 * 16 * 32, 32), oacc[2]);
    oacc[3] = wmma_bf16(up.v, load_b(Vc + 3 * 16 * 32, 32), oacc[3]);
    // all LDS reads of this buffer retired before anyone overwrites it
    asm volatile("s_wait_dscnt 0" ::: "memory");
  }

  float inv[8];
#pragma unroll
  for (int r = 0; r < 8; ++r) inv[r] = 1.0f / lrow[r];

#pragma unroll
  for (int h = 0; h < 4; ++h) {
    const int hd = h * 16 + n;
    const int f  = hd >> 1;
#pragma unroll
    for (int r = 0; r < 8; ++r) {
      const int t = t0 + mb + r;
      const float4 mm = *reinterpret_cast<const float4*>(mat_o + ((size_t)t * 32 + f) * 4);
      const float own  = oacc[h][r] * inv[r];
      const float part = __shfl_xor(own, 1);       // pair = adjacent column
      const float rot = (lane & 1) ? (mm.z * part + mm.w * own)
                                   : (mm.x * own + mm.y * part);
      Ot[((size_t)b * T_ + t) * C_ + hd * 8 + nh] = f2bf(rot);
    }
  }
}

// ---------------------------------------------------------------------------
// Kernel 3: out = w_o @ O + b_o  (fp32 output [B][C][T])
// ---------------------------------------------------------------------------
__global__ void __launch_bounds__(128) oproj_kernel(
    const unsigned short* __restrict__ Wo, const unsigned short* __restrict__ Ot,
    const float* __restrict__ bo, float* __restrict__ out)
{
  const int warp = threadIdx.x >> 5;
  const int lane = threadIdx.x & 31;
  const int wid  = blockIdx.x * 4 + warp;          // 8192 = 16 * 32 * 16
  const int ntile = wid & 15;
  const int mtile = (wid >> 4) & 31;
  const int b     = wid >> 9;
  const int o0   = mtile * 16;
  const int tok0 = ntile * 64;

  Acc4 acc = gemm16x64_k512(Wo + (size_t)o0 * C_,
                            Ot + ((size_t)b * T_ + tok0) * C_);
  v8f av[4] = {acc.a0, acc.a1, acc.a2, acc.a3};

  const int n = lane & 15, mb = (lane >> 4) << 3;
#pragma unroll
  for (int j = 0; j < 4; ++j) {
    const int t = tok0 + j * 16 + n;
#pragma unroll
    for (int r = 0; r < 8; ++r) {
      const int oo = o0 + mb + r;
      out[((size_t)b * C_ + oo) * T_ + t] = av[j][r] + bo[oo];
    }
  }
}

// ---------------------------------------------------------------------------
extern "C" void kernel_launch(void* const* d_in, const int* in_sizes, int n_in,
                              void* d_out, int out_size, void* d_ws, size_t ws_size,
                              hipStream_t stream) {
  (void)in_sizes; (void)n_in; (void)out_size; (void)ws_size;
  const float* x     = (const float*)d_in[0];
  const float* w_qkv = (const float*)d_in[1];
  const float* b_qkv = (const float*)d_in[2];
  const float* w_o   = (const float*)d_in[3];
  const float* b_o   = (const float*)d_in[4];
  const float* mat_q = (const float*)d_in[5];
  const float* mat_k = (const float*)d_in[6];
  const float* mat_v = (const float*)d_in[7];
  const float* mat_o = (const float*)d_in[8];
  float* out = (float*)d_out;

  char* ws = (char*)d_ws;
  unsigned short* Xt  = (unsigned short*)ws; ws += (size_t)B_ * T_ * C_ * 2;   // 16 MB
  unsigned short* Wq  = (unsigned short*)ws; ws += (size_t)3 * C_ * C_ * 2;    // 1.5 MB
  unsigned short* Wob = (unsigned short*)ws; ws += (size_t)C_ * C_ * 2;        // 0.5 MB
  unsigned short* Qb  = (unsigned short*)ws; ws += (size_t)B_ * NH * T_ * HD * 2;
  unsigned short* Kb  = (unsigned short*)ws; ws += (size_t)B_ * NH * T_ * HD * 2;
  unsigned short* Vt  = (unsigned short*)ws; ws += (size_t)B_ * NH * HD * T_ * 2;
  unsigned short* Ot  = (unsigned short*)ws; ws += (size_t)B_ * T_ * C_ * 2;

  prep_kernel<<<36864, 256, 0, stream>>>(x, w_qkv, w_o, Xt, Wq, Wob);
  qkv_kernel<<<6144, 128, 0, stream>>>(Wq, Xt, b_qkv, mat_q, mat_k, mat_v,
                                       Qb, Kb, Vt);
  attn_kernel<<<2048, 128, 0, stream>>>(Qb, Kb, Vt, mat_o, Ot);
  oproj_kernel<<<2048, 128, 0, stream>>>(Wob, Ot, b_o, out);
}
